// EVMPVaryVertCombine_78408922956214
// MI455X (gfx1250) — matrix operand order repaired and checked by reading, not compiled, not run on hardware
//
#include <hip/hip_runtime.h>

#define BN 64
#define NN 64
#define FD 64
#define STEPS 5

typedef __attribute__((ext_vector_type(16))) _Float16 v16h;
typedef __attribute__((ext_vector_type(4)))  _Float16 v4h;
typedef __attribute__((ext_vector_type(8)))  float    v8f;

union V16HU { v16h h; uint4 u[2]; };

// ---- fast transcendentals: prefer native V_TANH_F32 / V_RCP_F32 on gfx1250 ----
__device__ __forceinline__ float fast_rcp(float x) {
#if __has_builtin(__builtin_amdgcn_rcpf)
    return __builtin_amdgcn_rcpf(x);
#else
    return 1.0f / x;
#endif
}
__device__ __forceinline__ float fast_tanh(float x) {
#if __has_builtin(__builtin_amdgcn_tanhf)
    return __builtin_amdgcn_tanhf(x);
#else
    float e2 = __expf(2.0f * x);
    return 1.0f - 2.0f * fast_rcp(e2 + 1.0f);
#endif
}
__device__ __forceinline__ float fast_sigmoid(float x) {
#if __has_builtin(__builtin_amdgcn_tanhf)
    return 0.5f + 0.5f * __builtin_amdgcn_tanhf(0.5f * x);   // v_tanh_f32
#else
    return fast_rcp(1.0f + __expf(-x));
#endif
}

// WMMA D = A(16x32 f16) x B(32x16 f16) + C(16x16 f32)
__device__ __forceinline__ v8f wmma_(v16h a, v16h b, v8f c) {
    return __builtin_amdgcn_wmma_f32_16x16x32_f16(false, a, false, b, (short)0, c, false, false);
}

// B operand (32x16 slice of W^T): lane n<16 -> col n, K=chunk*32+{0..7,16..23} ;
// lane>=16 -> col n-16, K=chunk*32+{8..15,24..31}.  B[k,g] = W[g,k], W row-major f16 [64][64].
__device__ __forceinline__ v16h load_B(const _Float16* W, int nt, int lane, int chunk) {
    int g    = nt * 16 + (lane & 15);
    int koff = chunk * 32 + ((lane & 16) ? 8 : 0);
    const uint4* p = reinterpret_cast<const uint4*>(W + g * 64 + koff);
    V16HU r; r.u[0] = p[0]; r.u[1] = p[2];
    return r.h;
}

// A operand from an f32 LDS tile [16][64] (f32->f16 on the fly; used by small vertex kernel)
__device__ __forceinline__ v16h load_A_f32(const float* tile, int lane, int chunk) {
    int m  = lane & 15;
    int kb = (lane & 16) ? 8 : 0;
    const float* row = tile + m * 64 + chunk * 32 + kb;
    v16h a;
#pragma unroll
    for (int r = 0; r < 2; ++r) {
        const float4* p = reinterpret_cast<const float4*>(row + r * 16);
        float4 f0 = p[0], f1 = p[1];
        a[r*8+0] = (_Float16)f0.x; a[r*8+1] = (_Float16)f0.y;
        a[r*8+2] = (_Float16)f0.z; a[r*8+3] = (_Float16)f0.w;
        a[r*8+4] = (_Float16)f1.x; a[r*8+5] = (_Float16)f1.y;
        a[r*8+6] = (_Float16)f1.z; a[r*8+7] = (_Float16)f1.w;
    }
    return a;
}

// A-row fragment from an f16 LDS tile with 64-half row stride (two b128 ds loads).
__device__ __forceinline__ v16h load_Arow_f16(const _Float16* base, int row, int lane, int chunk) {
    int kb = (lane & 16) ? 8 : 0;
    const uint4* p = reinterpret_cast<const uint4*>(base + row * 64 + chunk * 32 + kb);
    V16HU r; r.u[0] = p[0]; r.u[1] = p[2];   // K runs at +0 and +16 halves
    return r.h;
}

// ---------------- prep: f32->f16 weights, zero e_v ----------------
__global__ void prep_kernel(const float* __restrict__ vW, const float* __restrict__ eW,
                            _Float16* __restrict__ vW16, _Float16* __restrict__ eW16,
                            float* __restrict__ e_v) {
    int idx = blockIdx.x * blockDim.x + threadIdx.x;
    if (idx < 6 * 64 * 64) {
        vW16[idx] = (_Float16)vW[idx];
        eW16[idx] = (_Float16)eW[idx];
    }
    if (idx < BN * NN * FD) e_v[idx] = 0.0f;
}

// ---------------- vertex GRU: v_h = GRU(e_v, v_h); v_h_up = v_h * vm ----------------
__global__ void __launch_bounds__(128)
vert_gru_kernel(const float* __restrict__ x_in,   // e_v  [B,N,F]
                const float* __restrict__ h_in,   // v_h  (step0: d_in v)
                float* __restrict__ h_out,        // ws v_h
                float* __restrict__ up_out,       // ws v_h_up
                const int* __restrict__ v_mask,   // [B,N]
                const _Float16* __restrict__ W16, // [6,64,64]
                const float* __restrict__ bvec,   // [6,64]
                float* out_tail, int last_step) {
    __shared__ float x_s[16 * FD];
    __shared__ float h_s[16 * FD];
    const int b    = blockIdx.x;
    const int tid  = threadIdx.x;
    const int lane = tid & 31;
    const int nt   = tid >> 5;          // wave id = N-tile
    const int hi8  = (lane & 16) ? 8 : 0;
    const int ccol = lane & 15;
    const int g    = nt * 16 + ccol;

    v16h Bw[6][2];
#pragma unroll
    for (int l = 0; l < 6; ++l)
#pragma unroll
        for (int c = 0; c < 2; ++c) Bw[l][c] = load_B(W16 + l * 4096, nt, lane, c);
    float bias[6];
#pragma unroll
    for (int l = 0; l < 6; ++l) bias[l] = bvec[l * 64 + g];

    for (int mt = 0; mt < 4; ++mt) {
        __syncthreads();
        const float4* xs = reinterpret_cast<const float4*>(x_in + ((size_t)b * NN + mt * 16) * FD);
        const float4* hs = reinterpret_cast<const float4*>(h_in + ((size_t)b * NN + mt * 16) * FD);
        reinterpret_cast<float4*>(x_s)[tid]       = xs[tid];
        reinterpret_cast<float4*>(x_s)[tid + 128] = xs[tid + 128];
        reinterpret_cast<float4*>(h_s)[tid]       = hs[tid];
        reinterpret_cast<float4*>(h_s)[tid + 128] = hs[tid + 128];
        __syncthreads();

        v16h xA0 = load_A_f32(x_s, lane, 0), xA1 = load_A_f32(x_s, lane, 1);
        v16h hA0 = load_A_f32(h_s, lane, 0), hA1 = load_A_f32(h_s, lane, 1);

        v8f a0 = {}, a1 = {}, a2 = {}, a3 = {}, a4 = {}, a5 = {};
        a0 = wmma_(xA0, Bw[0][0], a0); a0 = wmma_(xA1, Bw[0][1], a0);
        a1 = wmma_(hA0, Bw[1][0], a1); a1 = wmma_(hA1, Bw[1][1], a1);
        a2 = wmma_(xA0, Bw[2][0], a2); a2 = wmma_(xA1, Bw[2][1], a2);
        a3 = wmma_(hA0, Bw[3][0], a3); a3 = wmma_(hA1, Bw[3][1], a3);
        a4 = wmma_(xA0, Bw[4][0], a4); a4 = wmma_(xA1, Bw[4][1], a4);
        a5 = wmma_(hA0, Bw[5][0], a5); a5 = wmma_(hA1, Bw[5][1], a5);

        const int* vmp = v_mask + b * NN + mt * 16 + hi8;
        float* po = h_out  + ((size_t)b * NN + mt * 16) * FD;
        float* pu = up_out + ((size_t)b * NN + mt * 16) * FD;
#pragma unroll
        for (int v = 0; v < 8; ++v) {
            int   m    = v + hi8;
            float hold = h_s[m * FD + g];
            float r = fast_sigmoid(a0[v] + bias[0] + a1[v] + bias[1]);
            float z = fast_sigmoid(a2[v] + bias[2] + a3[v] + bias[3]);
            float n = fast_tanh(a4[v] + bias[4] + r * (a5[v] + bias[5]));
            float hn = (1.0f - z) * n + z * hold;
            float up = (vmp[v] > 0) ? hn : 0.0f;
            po[m * FD + g] = hn;
            pu[m * FD + g] = up;
            if (last_step) out_tail[((size_t)b * NN + mt * 16 + m) * FD + g] = up;
        }
    }
}

// ---- edge GRU + fused masked max over i:  block = (b, j-tile), loops i = 0..63 ----
__global__ void __launch_bounds__(128)
edge_gru_kernel(const float* eh_in,               // [B,N,N,F] (step0: d_in e, else d_out)
                float* eh_out,                    // d_out e region (in-place for s>=1)
                const float* __restrict__ vh_up,  // ws [B,N,F]
                const int* __restrict__ e_mask,   // [B,N,N]
                const _Float16* __restrict__ W16, // [6,64,64]
                const float* __restrict__ bvec,   // [6,64]
                float* __restrict__ e_v,          // ws [B,N,F]
                int last_step) {
    __shared__ _Float16 vh_s[NN * FD];  // 8 KB f16 copy of v_h_up[b]
    __shared__ float    h_s[16 * FD];   // 4 KB staged e_h tile (f32, for hold gather)
    __shared__ _Float16 hf_s[16 * FD];  // 2 KB f16 copy of staged tile (for A fragments)
    const int b    = blockIdx.x >> 2;
    const int jt   = blockIdx.x & 3;
    const int j0   = jt * 16;
    const int tid  = threadIdx.x;
    const int lane = tid & 31;
    const int nt   = tid >> 5;
    const int hi8  = (lane & 16) ? 8 : 0;
    const int ccol = lane & 15;
    const int g    = nt * 16 + ccol;

    {   // stage v_h_up[b] -> f16 LDS
        const float4* src = reinterpret_cast<const float4*>(vh_up + (size_t)b * NN * FD);
        for (int idx = tid; idx < (NN * FD) / 4; idx += 128) {
            float4 f = src[idx];
            v4h hv; hv[0] = (_Float16)f.x; hv[1] = (_Float16)f.y;
            hv[2] = (_Float16)f.z; hv[3] = (_Float16)f.w;
            *reinterpret_cast<v4h*>(vh_s + idx * 4) = hv;
        }
    }
    __syncthreads();

    v16h Bw[6][2];
#pragma unroll
    for (int l = 0; l < 6; ++l)
#pragma unroll
        for (int c = 0; c < 2; ++c) Bw[l][c] = load_B(W16 + l * 4096, nt, lane, c);
    float bias[6];
#pragma unroll
    for (int l = 0; l < 6; ++l) bias[l] = bvec[l * 64 + g];

    // j-side halves of the outer product (constant across i)
    v16h jh0 = load_Arow_f16(vh_s, j0 + (lane & 15), lane, 0);
    v16h jh1 = load_Arow_f16(vh_s, j0 + (lane & 15), lane, 1);

    float vmax[8];
#pragma unroll
    for (int v = 0; v < 8; ++v) vmax[v] = -3.0e38f;

    // register double-buffer for the 4 KB e_h tile: 2 float4 per thread
    float4 r0, r1;
    {
        const float4* src =
            reinterpret_cast<const float4*>(eh_in + (((size_t)b * NN + 0) * NN + j0) * FD);
        r0 = src[tid]; r1 = src[tid + 128];
    }

    for (int i = 0; i < NN; ++i) {
        __syncthreads();   // previous iteration's LDS reads complete
        // commit prefetched tile to LDS (f32 + f16 copies)
        reinterpret_cast<float4*>(h_s)[tid]       = r0;
        reinterpret_cast<float4*>(h_s)[tid + 128] = r1;
        {
            v4h ha, hb;
            ha[0] = (_Float16)r0.x; ha[1] = (_Float16)r0.y;
            ha[2] = (_Float16)r0.z; ha[3] = (_Float16)r0.w;
            hb[0] = (_Float16)r1.x; hb[1] = (_Float16)r1.y;
            hb[2] = (_Float16)r1.z; hb[3] = (_Float16)r1.w;
            *reinterpret_cast<v4h*>(hf_s + tid * 4)         = ha;
            *reinterpret_cast<v4h*>(hf_s + (tid + 128) * 4) = hb;
        }
        // start next tile's global loads (overlaps with this iteration's compute)
        float4 n0, n1;
        if (i + 1 < NN) {
            const float4* src =
                reinterpret_cast<const float4*>(eh_in + (((size_t)b * NN + i + 1) * NN + j0) * FD);
            n0 = src[tid]; n1 = src[tid + 128];
        }
        __syncthreads();   // tile visible to all waves

        v16h hA0 = load_Arow_f16(hf_s, lane & 15, lane, 0);
        v16h hA1 = load_Arow_f16(hf_s, lane & 15, lane, 1);
        v16h ih0 = load_Arow_f16(vh_s, i, lane, 0);
        v16h ih1 = load_Arow_f16(vh_s, i, lane, 1);
        v16h xA0 = jh0 * ih0;   // v_pk_mul_f16 outer-product rows
        v16h xA1 = jh1 * ih1;

        v8f a0 = {}, a1 = {}, a2 = {}, a3 = {}, a4 = {}, a5 = {};
        a0 = wmma_(xA0, Bw[0][0], a0); a0 = wmma_(xA1, Bw[0][1], a0);
        a1 = wmma_(hA0, Bw[1][0], a1); a1 = wmma_(hA1, Bw[1][1], a1);
        a2 = wmma_(xA0, Bw[2][0], a2); a2 = wmma_(xA1, Bw[2][1], a2);
        a3 = wmma_(hA0, Bw[3][0], a3); a3 = wmma_(hA1, Bw[3][1], a3);
        a4 = wmma_(xA0, Bw[4][0], a4); a4 = wmma_(xA1, Bw[4][1], a4);
        a5 = wmma_(hA0, Bw[5][0], a5); a5 = wmma_(hA1, Bw[5][1], a5);

        const int4* emp =
            reinterpret_cast<const int4*>(e_mask + ((size_t)(b * NN + i) * NN + j0 + hi8));
        int4 em0 = emp[0], em1 = emp[1];
        int emv[8] = {em0.x, em0.y, em0.z, em0.w, em1.x, em1.y, em1.z, em1.w};

        float* outp = eh_out + (((size_t)b * NN + i) * NN + j0) * FD;
#pragma unroll
        for (int v = 0; v < 8; ++v) {
            int   m    = v + hi8;
            float hold = h_s[m * FD + g];
            float r = fast_sigmoid(a0[v] + bias[0] + a1[v] + bias[1]);
            float z = fast_sigmoid(a2[v] + bias[2] + a3[v] + bias[3]);
            float n = fast_tanh(a4[v] + bias[4] + r * (a5[v] + bias[5]));
            float hn = (1.0f - z) * n + z * hold;
            float mv = (emv[v] > 0) ? hn : 0.0f;
            vmax[v]  = fmaxf(vmax[v], mv);
            outp[m * FD + g] = last_step ? mv : hn;
        }
        r0 = n0; r1 = n1;
    }

    float* evp = e_v + ((size_t)b * NN + j0) * FD + g;
#pragma unroll
    for (int v = 0; v < 8; ++v) evp[(v + hi8) * FD] = vmax[v];
}

extern "C" void kernel_launch(void* const* d_in, const int* in_sizes, int n_in,
                              void* d_out, int out_size, void* d_ws, size_t ws_size,
                              hipStream_t stream) {
    const float* e      = (const float*)d_in[0];
    const float* v      = (const float*)d_in[1];
    const int*   e_mask = (const int*)d_in[2];
    const int*   v_mask = (const int*)d_in[3];
    const float* v_W    = (const float*)d_in[4];
    const float* v_b    = (const float*)d_in[5];
    const float* e_W    = (const float*)d_in[6];
    const float* e_b    = (const float*)d_in[7];

    const size_t E_ELEMS = (size_t)BN * NN * NN * FD;  // 16,777,216
    const size_t V_ELEMS = (size_t)BN * NN * FD;       // 262,144

    float* out_e = (float*)d_out;            // e_h state lives here in-place
    float* out_v = out_e + E_ELEMS;          // v_h_up tail

    char* ws = (char*)d_ws;
    float*    vh   = (float*)ws;                 ws += V_ELEMS * sizeof(float);
    float*    vhup = (float*)ws;                 ws += V_ELEMS * sizeof(float);
    float*    ev   = (float*)ws;                 ws += V_ELEMS * sizeof(float);
    _Float16* vW16 = (_Float16*)ws;              ws += 6 * 64 * 64 * sizeof(_Float16);
    _Float16* eW16 = (_Float16*)ws;

    prep_kernel<<<1024, 256, 0, stream>>>(v_W, e_W, vW16, eW16, ev);

    for (int s = 0; s < STEPS; ++s) {
        int last = (s == STEPS - 1) ? 1 : 0;
        vert_gru_kernel<<<64, 128, 0, stream>>>(
            ev, (s == 0) ? v : vh, vh, vhup, v_mask, vW16, v_b, out_v, last);
        edge_gru_kernel<<<256, 128, 0, stream>>>(
            (s == 0) ? e : out_e, out_e, vhup, e_mask, eW16, e_b, ev, last);
    }
}
